// MCRM_34385508171907
// MI455X (gfx1250) — matrix operand order repaired and checked by reading, not compiled, not run on hardware
//
#include <hip/hip_runtime.h>
#include <cstdint>

typedef __bf16 bf16;
typedef __attribute__((ext_vector_type(16))) __bf16 v16bf;
typedef __attribute__((ext_vector_type(8)))  __bf16 v8bf;
typedef __attribute__((ext_vector_type(8)))  float  v8f;
typedef __attribute__((ext_vector_type(8)))  unsigned int v8u;
typedef __attribute__((ext_vector_type(4)))  unsigned int v4u;

#define WMMA_BF16(a,b,c) __builtin_amdgcn_wmma_f32_16x16x32_bf16(false,(a),false,(b),(short)0,(c),false,false)

static constexpr int Dc   = 384;
static constexpr int HWp  = 96;
static constexpr int Ltok = 9216;   // 96*96
static constexpr int Skv  = 756;
static constexpr int Spad = 768;
static constexpr int DHd  = 48;

static __device__ __forceinline__ bf16 f2b(float f) {
  unsigned u = __float_as_uint(f);
  unsigned r = (u + 0x7FFFu + ((u >> 16) & 1u)) >> 16;
  return __builtin_bit_cast(bf16, (unsigned short)r);
}
static __device__ __forceinline__ v8bf zero8() {
  v4u z = {0u,0u,0u,0u};
  return __builtin_bit_cast(v8bf, z);
}
static __device__ __forceinline__ v16bf zero16() {
  v8u z = {0u,0u,0u,0u,0u,0u,0u,0u};
  return __builtin_bit_cast(v16bf, z);
}
static __device__ __forceinline__ v16bf cat8(v8bf lo, v8bf hi) {
  return __builtin_shufflevector(lo, hi, 0,1,2,3,4,5,6,7,8,9,10,11,12,13,14,15);
}
static __device__ __forceinline__ v8f zero8f() {
  v8f z = {0.f,0.f,0.f,0.f,0.f,0.f,0.f,0.f};
  return z;
}
static __device__ __forceinline__ float geluf(float x) {
  return 0.5f * x * (1.f + erff(x * 0.70710678118654752f));
}
// A-matrix fragment (16x32, bf16): lane half h, row lane&15.
// half 0: elems 0..7 = K k..k+7,  elems 8..15 = K k+16..k+23
// half 1: elems 0..7 = K k+8..15, elems 8..15 = K k+24..k+31
static __device__ __forceinline__ v16bf loadA(const bf16* row, int k, int half) {
  v8bf lo = *(const v8bf*)(row + k + half * 8);
  v8bf hi = *(const v8bf*)(row + k + 16 + half * 8);
  return cat8(lo, hi);
}
// B-matrix fragment (32x16): column lane&15; same K order as loadA (half h: k+8h..+7, k+16+8h..+7)
static __device__ __forceinline__ v16bf loadB(const bf16* row, int k, int half) {
  return cat8(*(const v8bf*)(row + k + half * 8), *(const v8bf*)(row + k + 16 + half * 8));
}

// ---------------------------------------------------------------- utility
__global__ void __launch_bounds__(256)
k_f32_to_bf16(const float* __restrict__ in, bf16* __restrict__ out, int n) {
  int i = blockIdx.x * 256 + threadIdx.x;
  if (i < n) out[i] = f2b(in[i]);
}

// ---------------------------------------------------------------- saliency map
__global__ void __launch_bounds__(256)
k_tam(const float* __restrict__ x, const float* __restrict__ sw,
      const float* __restrict__ sb, float* __restrict__ tam) {
  int p = blockIdx.x * 256 + threadIdx.x;
  if (p >= Ltok) return;
  const float* g = x + (size_t)4 * Dc * Ltok + p;
  float s = 0.f;
  for (int c = 0; c < Dc; ++c) s += sw[c] * g[(size_t)c * Ltok];
  s += sb[0];
  tam[p] = 1.f / (1.f + __expf(-s));
}

// ------------------------- scale loc by tam patch + transpose to (n,l,c) f32/bf16
__global__ void __launch_bounds__(256)
k_scale_pack(const float* __restrict__ x, const float* __restrict__ tam,
             float* __restrict__ locres, bf16* __restrict__ qin) {
  __shared__ float tile[32][33];
  int n = blockIdx.z;
  int l0 = blockIdx.x * 32, c0 = blockIdx.y * 32;
  int tx = threadIdx.x, ty = threadIdx.y;
  int a = n >> 1, b = n & 1;
  int l = l0 + tx;
  int y = l / HWp, xp = l % HWp;
  float tv = tam[(a * 48 + (y >> 1)) * HWp + (b * 48 + (xp >> 1))];
  for (int i = 0; i < 4; ++i) {
    int c = c0 + ty + 8 * i;
    tile[ty + 8 * i][tx] = x[((size_t)n * Dc + c) * Ltok + l] * tv;
  }
  __syncthreads();
  for (int i = 0; i < 4; ++i) {
    int c  = c0 + tx;
    int lw = l0 + ty + 8 * i;
    float v = tile[tx][ty + 8 * i];
    size_t o = ((size_t)n * Ltok + lw) * Dc + c;
    locres[o] = v;
    qin[o] = f2b(v);
  }
}

// ------------------------------------------------- pyramid pooling -> kv (n,s,c) bf16
__global__ void __launch_bounds__(256)
k_pool(const float* __restrict__ x, bf16* __restrict__ kvin) {
  int idx = blockIdx.x * 256 + threadIdx.x;
  if (idx >= 4 * Skv * Dc) return;
  int c = idx % Dc;
  int t = idx / Dc;
  int s = t % Skv;
  int p = t / Skv;
  int ty, tx, kk;
  if (s < 576)      { ty = s / 24;          tx = s % 24;          kk = 2; }
  else if (s < 720) { int s2 = s - 576; ty = s2 / 12; tx = s2 % 12; kk = 4; }
  else              { int s2 = s - 720; ty = s2 / 6;  tx = s2 % 6;  kk = 8; }
  const float* g = x + ((size_t)4 * Dc + c) * Ltok;
  int by = (p >> 1) * 48 + ty * kk, bx = (p & 1) * 48 + tx * kk;
  float sum = 0.f;
  for (int i = 0; i < kk; ++i)
    for (int j = 0; j < kk; ++j)
      sum += g[(by + i) * HWp + (bx + j)];
  kvin[((size_t)p * Spad + s) * Dc + c] = f2b(sum / (float)(kk * kk));
}

// ------------------------------------------------- generic WMMA GEMM: C = A*B^T + bias
// A: (M,K) bf16 row-major.  B: (N,K) bf16 row-major (weights).
// mode 0: Cf f32 (M,N); mode 1: Cb bf16 (M,N); mode 2: Cb bf16 transposed (N, ldT)
// Out-of-range M rows are handled by CLAMPING the A row (keeps the inner loop
// branch-free; garbage accumulates only in D rows whose stores are skipped).
__global__ void __launch_bounds__(128)
k_gemm(const bf16* __restrict__ A, const bf16* __restrict__ B, const float* __restrict__ bias,
       float* __restrict__ Cf, bf16* __restrict__ Cb,
       int M, int N, int K,
       long long aStride, long long bStride, long long biasStride, long long cStride,
       int act, int mode, int ldT)
{
  int Nt2 = N >> 5;
  int Mt2 = (M + 31) >> 5;
  int batch = blockIdx.y;
  int wid = blockIdx.x * 4 + (int)(threadIdx.x >> 5);
  if (wid >= Mt2 * Nt2) return;
  int tM = (wid / Nt2) << 5;
  int tN = (wid % Nt2) << 5;
  int lane = threadIdx.x & 31;
  int r = lane & 15, half = lane >> 4;

  const bf16* Ab = A + (size_t)batch * aStride;
  const bf16* Bb = B + (size_t)batch * bStride;
  int row0 = min(tM + r, M - 1);        // clamped: branch-free inner loop
  int row1 = min(tM + 16 + r, M - 1);
  const bf16* pa0 = Ab + (size_t)row0 * K + half * 16;
  const bf16* pa1 = Ab + (size_t)row1 * K + half * 16;
  const bf16* pb0 = Bb + (size_t)(tN + r) * K + half * 16;
  const bf16* pb1 = Bb + (size_t)(tN + 16 + r) * K + half * 16;

  v8f acc00 = zero8f(), acc01 = zero8f(), acc10 = zero8f(), acc11 = zero8f();
  for (int k = 0; k < K; k += 32) {
    v16bf a0 = cat8(*(const v8bf*)pa0, *(const v8bf*)(pa0 + 8));
    v16bf a1 = cat8(*(const v8bf*)pa1, *(const v8bf*)(pa1 + 8));
    v16bf b0 = *(const v16bf*)pb0;
    v16bf b1 = *(const v16bf*)pb1;
    acc00 = WMMA_BF16(a0, b0, acc00);
    acc01 = WMMA_BF16(a0, b1, acc01);
    acc10 = WMMA_BF16(a1, b0, acc10);
    acc11 = WMMA_BF16(a1, b1, acc11);
    pa0 += 32; pa1 += 32; pb0 += 32; pb1 += 32;
  }
  float bv0 = bias ? bias[(size_t)batch * biasStride + tN + r] : 0.f;
  float bv1 = bias ? bias[(size_t)batch * biasStride + tN + 16 + r] : 0.f;

  auto store = [&](v8f acc, int rowBase, int col, float bv) {
    for (int e = 0; e < 8; ++e) {
      int row = rowBase + 8 * half + e;
      if (row >= M) continue;
      float v = acc[e] + bv;
      if (act == 1) v = geluf(v);
      if (mode == 0)      Cf[(size_t)batch * cStride + (size_t)row * N + col] = v;
      else if (mode == 1) Cb[(size_t)batch * cStride + (size_t)row * N + col] = f2b(v);
      else                Cb[(size_t)batch * cStride + (size_t)col * ldT + row] = f2b(v);
    }
  };
  store(acc00, tM,      tN + r,      bv0);
  store(acc01, tM,      tN + 16 + r, bv1);
  store(acc10, tM + 16, tN + r,      bv0);
  store(acc11, tM + 16, tN + 16 + r, bv1);
}

// ------------------------------------------------- attention: one wave per (qtile, head, n)
__global__ void __launch_bounds__(32)
k_attn(const bf16* __restrict__ Q, const bf16* __restrict__ Kp, const bf16* __restrict__ Vt,
       bf16* __restrict__ AO)
{
  __shared__ __align__(32) float sc[16 * Spad];
  bf16* pbuf = (bf16*)sc;   // P row m occupies first Spad bf16 slots of slot m*2*Spad

  int qt = blockIdx.x, head = blockIdx.y, n = blockIdx.z;
  int lane = threadIdx.x & 31;
  int r = lane & 15, half = lane >> 4;

  const bf16* qrow = Q + ((size_t)n * Ltok + qt * 16 + r) * Dc + head * DHd;
  v16bf qa0 = loadA(qrow, 0, half);                        // head feats 0..31
  v8bf  ql1 = *(const v8bf*)(qrow + 32 + half * 8);        // feats 32..47, rest zero-pad
  v16bf qa1 = cat8(ql1, zero8());

  const float scale = 0.14433756729740643f;  // 1/sqrt(48)
  for (int st = 0; st < Spad / 16; ++st) {
    const bf16* krow = Kp + ((size_t)n * Spad + st * 16 + r) * Dc + head * DHd;
    v16bf b0 = loadB(krow, 0, half);
    v16bf b1 = cat8(*(const v8bf*)(krow + 32 + half * 8), zero8());
    v8f acc = zero8f();
    acc = WMMA_BF16(qa0, b0, acc);
    acc = WMMA_BF16(qa1, b1, acc);
    int s = st * 16 + r;
    bool valid = s < Skv;
    for (int e = 0; e < 8; ++e)
      sc[(8 * half + e) * Spad + s] = valid ? acc[e] * scale : -1e30f;
  }
  __syncthreads();

  {  // softmax: lane pair (r, r+16) shares row r; halves cover cols [0,384)/[384,768)
    float* base = sc + r * Spad + half * 384;
    float mx = -1e30f;
    for (int j = 0; j < 384; ++j) mx = fmaxf(mx, base[j]);
    mx = fmaxf(mx, __shfl_xor(mx, 16));
    float sum = 0.f;
    for (int j = 0; j < 384; ++j) sum += __expf(base[j] - mx);
    sum += __shfl_xor(sum, 16);
    float inv = 1.f / sum;
    bf16* pb = pbuf + r * (2 * Spad) + half * 384;  // in-place, stays behind f32 reads
    for (int j = 0; j < 384; ++j) pb[j] = f2b(__expf(base[j] - mx) * inv);
  }
  __syncthreads();

  const bf16* pr = pbuf + r * (2 * Spad);
  for (int nt = 0; nt < 3; ++nt) {
    int feat = head * DHd + nt * 16 + r;
    const bf16* vrow = Vt + ((size_t)n * Dc + feat) * Spad;
    v8f acc = zero8f();
    for (int kt = 0; kt < Spad; kt += 32) {
      v16bf a = loadA(pr, kt, half);
      v16bf b = loadB(vrow, kt, half);
      acc = WMMA_BF16(a, b, acc);
    }
    for (int e = 0; e < 8; ++e) {
      int row = qt * 16 + 8 * half + e;
      AO[((size_t)n * Ltok + row) * Dc + feat] = f2b(acc[e]);
    }
  }
}

// ------------------------------------------------- residual + layernorm (wave per token)
__global__ void __launch_bounds__(256)
k_ln(const float* __restrict__ A, const float* __restrict__ Bs,
     const float* __restrict__ g, const float* __restrict__ be,
     float* __restrict__ outF, bf16* __restrict__ outB, int nTok)
{
  int wave = blockIdx.x * 8 + (int)(threadIdx.x >> 5);
  if (wave >= nTok) return;
  int lane = threadIdx.x & 31;
  const float* pa = A + (size_t)wave * Dc;
  const float* pb = Bs + (size_t)wave * Dc;
  float vals[12];
  float s = 0.f;
  for (int i = 0; i < 12; ++i) {
    float v = pa[lane + i * 32] + pb[lane + i * 32];
    vals[i] = v; s += v;
  }
  for (int o = 16; o; o >>= 1) s += __shfl_xor(s, o);
  float mean = s * (1.f / 384.f);
  float vs = 0.f;
  for (int i = 0; i < 12; ++i) { float d = vals[i] - mean; vs += d * d; }
  for (int o = 16; o; o >>= 1) vs += __shfl_xor(vs, o);
  float rstd = rsqrtf(vs * (1.f / 384.f) + 1e-5f);
  for (int i = 0; i < 12; ++i) {
    int c = lane + i * 32;
    float yv = (vals[i] - mean) * rstd * g[c] + be[c];
    if (outF) outF[(size_t)wave * Dc + c] = yv;
    if (outB) outB[(size_t)wave * Dc + c] = f2b(yv);
  }
}

// ------------------------------------------------- (n,l,c) -> (n,c,l) into d_out[0..4)
__global__ void __launch_bounds__(256)
k_out_src(const float* __restrict__ srcF, float* __restrict__ out) {
  __shared__ float tile[32][33];
  int n = blockIdx.z;
  int l0 = blockIdx.x * 32, c0 = blockIdx.y * 32;
  int tx = threadIdx.x, ty = threadIdx.y;
  for (int i = 0; i < 4; ++i)
    tile[ty + 8 * i][tx] = srcF[((size_t)n * Ltok + l0 + ty + 8 * i) * Dc + c0 + tx];
  __syncthreads();
  for (int i = 0; i < 4; ++i)
    out[((size_t)n * Dc + c0 + ty + 8 * i) * Ltok + l0 + tx] = tile[tx][ty + 8 * i];
}

// ------------------------------------------------- glb update into d_out[4]
__global__ void __launch_bounds__(256)
k_out_glb(const float* __restrict__ xin, const float* __restrict__ srcF, float* __restrict__ out) {
  int idx = blockIdx.x * 256 + threadIdx.x;
  if (idx >= Dc * Ltok) return;
  int pix = idx % Ltok;
  int c = idx / Ltok;
  int y = pix / HWp, xx = pix % HWp;
  int a = y / 48, b = xx / 48;
  int p = 2 * a + b;
  int yy = 2 * (y - 48 * a), xq = 2 * (xx - 48 * b);
  float v = xin[((size_t)4 * Dc + c) * Ltok + pix] +
            srcF[((size_t)p * Ltok + yy * HWp + xq) * Dc + c];
  out[((size_t)4 * Dc + c) * Ltok + pix] = v;
}

// ================================================================ host
extern "C" void kernel_launch(void* const* d_in, const int* in_sizes, int n_in,
                              void* d_out, int out_size, void* d_ws, size_t ws_size,
                              hipStream_t stream) {
  (void)in_sizes; (void)n_in; (void)out_size; (void)ws_size;
  const float* x          = (const float*)d_in[0];
  const float* sal_w      = (const float*)d_in[1];
  const float* sal_b      = (const float*)d_in[2];
  const float* attn_in_w  = (const float*)d_in[3];
  const float* attn_in_b  = (const float*)d_in[4];
  const float* attn_out_w = (const float*)d_in[5];
  const float* attn_out_b = (const float*)d_in[6];
  const float* w3 = (const float*)d_in[7];
  const float* b3 = (const float*)d_in[8];
  const float* w4 = (const float*)d_in[9];
  const float* b4 = (const float*)d_in[10];
  const float* g1 = (const float*)d_in[11];
  const float* be1 = (const float*)d_in[12];
  const float* g2 = (const float*)d_in[13];
  const float* be2 = (const float*)d_in[14];
  float* out = (float*)d_out;

  char* ws = (char*)d_ws;
  size_t off = 0;
  auto alloc = [&](size_t bytes) -> void* {
    void* p = ws + off;
    off += (bytes + 255) & ~(size_t)255;
    return p;
  };
  const size_t TOKB = (size_t)4 * Ltok * Dc;     // 14,155,776 elements
  float* tam    = (float*)alloc((size_t)Ltok * 4);
  float* locres = (float*)alloc(TOKB * 4);                 // also srcF after LN2
  bf16*  qin    = (bf16*)alloc(TOKB * 2);
  bf16*  qproj  = (bf16*)alloc(TOKB * 2);
  bf16*  hid    = qin;                                     // alias: qin+qproj = 36864x768 bf16
  bf16*  kvin   = (bf16*)alloc((size_t)4 * Spad * Dc * 2);
  bf16*  wib    = (bf16*)alloc((size_t)4 * 1152 * 384 * 2);
  bf16*  wob    = (bf16*)alloc((size_t)4 * 384 * 384 * 2);
  bf16*  w3b    = (bf16*)alloc((size_t)768 * 384 * 2);
  bf16*  w4b    = (bf16*)alloc((size_t)384 * 768 * 2);
  bf16*  kproj  = (bf16*)alloc((size_t)4 * Spad * Dc * 2);
  bf16*  vt     = (bf16*)alloc((size_t)4 * Dc * Spad * 2);
  bf16*  ao     = (bf16*)alloc(TOKB * 2);
  float* aproj  = (float*)alloc(TOKB * 4);                 // also ff after LN1
  float* ff     = aproj;
  float* srcf   = (float*)alloc(TOKB * 4);
  bf16*  srcb   = (bf16*)alloc(TOKB * 2);
  float* src2   = locres;

  // weight conversion f32 -> bf16
  {
    int n1 = 4 * 1152 * 384;
    k_f32_to_bf16<<<(n1 + 255) / 256, 256, 0, stream>>>(attn_in_w, wib, n1);
    int n2 = 4 * 384 * 384;
    k_f32_to_bf16<<<(n2 + 255) / 256, 256, 0, stream>>>(attn_out_w, wob, n2);
    int n3 = 768 * 384;
    k_f32_to_bf16<<<(n3 + 255) / 256, 256, 0, stream>>>(w3, w3b, n3);
    k_f32_to_bf16<<<(n3 + 255) / 256, 256, 0, stream>>>(w4, w4b, n3);
  }

  k_tam<<<Ltok / 256, 256, 0, stream>>>(x, sal_w, sal_b, tam);
  k_scale_pack<<<dim3(Ltok / 32, Dc / 32, 4), dim3(32, 8), 0, stream>>>(x, tam, locres, qin);
  k_pool<<<(4 * Skv * Dc + 255) / 256, 256, 0, stream>>>(x, kvin);

  auto gemm = [&](const bf16* A, const bf16* B, const float* bias, float* Cf, bf16* Cb,
                  int M, int N, int K, int batch,
                  long long aS, long long bS, long long biS, long long cS,
                  int act, int mode, int ldT) {
    int mt = (M + 31) / 32, nt = N / 32;
    int blocks = (mt * nt + 3) / 4;
    k_gemm<<<dim3(blocks, batch), 128, 0, stream>>>(A, B, bias, Cf, Cb, M, N, K,
                                                    aS, bS, biS, cS, act, mode, ldT);
  };

  // Q projection: (n,9216,384) x wq[n] -> qproj bf16
  gemm(qin, wib, attn_in_b, nullptr, qproj, Ltok, Dc, Dc, 4,
       (long long)Ltok * Dc, (long long)1152 * 384, 1152, (long long)Ltok * Dc, 0, 1, 0);
  // K projection: (n,756,384) -> kproj bf16 (padded rows masked later)
  gemm(kvin, wib + (size_t)384 * 384, attn_in_b + 384, nullptr, kproj, Skv, Dc, Dc, 4,
       (long long)Spad * Dc, (long long)1152 * 384, 1152, (long long)Spad * Dc, 0, 1, 0);
  // V projection, stored transposed: vt (n, 384, 768)
  gemm(kvin, wib + (size_t)2 * 384 * 384, attn_in_b + 768, nullptr, vt, Skv, Dc, Dc, 4,
       (long long)Spad * Dc, (long long)1152 * 384, 1152, (long long)Dc * Spad, 0, 2, Spad);

  k_attn<<<dim3(Ltok / 16, 8, 4), 32, 0, stream>>>(qproj, kproj, vt, ao);

  // output projection -> f32
  gemm(ao, wob, attn_out_b, aproj, nullptr, Ltok, Dc, Dc, 4,
       (long long)Ltok * Dc, (long long)384 * 384, 384, (long long)Ltok * Dc, 0, 0, 0);

  k_ln<<<(4 * Ltok) / 8, 256, 0, stream>>>(locres, aproj, g1, be1, srcf, srcb, 4 * Ltok);

  // FFN1: 36864x384 x (768x384)^T, GELU, bf16 out
  gemm(srcb, w3b, b3, nullptr, hid, 4 * Ltok, 768, 384, 1, 0, 0, 0, 0, 1, 1, 0);
  // FFN2: 36864x768 x (384x768)^T, f32 out
  gemm(hid, w4b, b4, ff, nullptr, 4 * Ltok, 384, 768, 1, 0, 0, 0, 0, 0, 0, 0);

  k_ln<<<(4 * Ltok) / 8, 256, 0, stream>>>(srcf, ff, g2, be2, src2, nullptr, 4 * Ltok);

  k_out_src<<<dim3(Ltok / 32, Dc / 32, 4), dim3(32, 8), 0, stream>>>(src2, out);
  k_out_glb<<<(Dc * Ltok + 255) / 256, 256, 0, stream>>>(x, src2, out);
}
